// ProgramModel_27676769255908
// MI455X (gfx1250) — matrix operand order repaired and checked by reading, not compiled, not run on hardware
//
#include <hip/hip_runtime.h>
#include <hip/hip_bf16.h>

// ---------------- problem constants ----------------
#define BB 2
#define TT 2048
#define DD 1024
#define LL 8
#define NPAT 256
#define PDIM 32
#define TOPK 4
#define VV 32000
#define NTOK (BB * TT)          // 4096
#define EPS_RMS 1.1920929e-07f

// ---------------- vector types ----------------
typedef __attribute__((ext_vector_type(16))) __bf16          v16bf;
typedef __attribute__((ext_vector_type(8)))  float           v8f;
typedef __attribute__((ext_vector_type(16))) unsigned short  v16u;
typedef __attribute__((ext_vector_type(8)))  unsigned short  ushort8;
typedef __attribute__((ext_vector_type(4)))  unsigned int    u32x4;
typedef __attribute__((ext_vector_type(8)))  int             i32x8;
typedef __attribute__((ext_vector_type(4)))  int             i32x4;

// round-to-nearest-even float -> bf16 bits
static __device__ __forceinline__ unsigned short f2bf(float f) {
    unsigned int u = __float_as_uint(f);
    unsigned int r = u + 0x7FFFu + ((u >> 16) & 1u);
    return (unsigned short)(r >> 16);
}

// ---------------- 1) embedding gather ----------------
__global__ __launch_bounds__(256) void embed_kernel(
    const int* __restrict__ tokens, const float* __restrict__ embed,
    float* __restrict__ x)
{
    int t = blockIdx.x;
    int tok = tokens[t];
    const float4* src = (const float4*)(embed + (size_t)tok * DD);
    float4* dst = (float4*)(x + (size_t)t * DD);
    for (int i = threadIdx.x; i < DD / 4; i += blockDim.x) dst[i] = src[i];
}

// ---------------- 2) hasher + sim + top4 + softmax (one wave32 per token) --
__global__ __launch_bounds__(256) void hasher_topk_kernel(
    const float* __restrict__ x, const float* __restrict__ hw /*[32,1024]*/,
    const float* __restrict__ keys /*[256,32]*/,
    int* __restrict__ tidx, float* __restrict__ tw)
{
    int lane = threadIdx.x & 31;
    int wave = threadIdx.x >> 5;
    int t = blockIdx.x * 8 + wave;

    const float4* x4 = (const float4*)(x + (size_t)t * DD);
    const float4* h4 = (const float4*)(hw + (size_t)lane * DD);
    float h = 0.f;
    #pragma unroll 4
    for (int i = 0; i < DD / 4; ++i) {
        float4 a = x4[i], b = h4[i];
        h = fmaf(a.x, b.x, fmaf(a.y, b.y, fmaf(a.z, b.z, fmaf(a.w, b.w, h))));
    }

    float sim[8];
    #pragma unroll
    for (int g = 0; g < 8; ++g) sim[g] = 0.f;
    for (int p = 0; p < PDIM; ++p) {
        float hp = __shfl(h, p, 32);
        #pragma unroll
        for (int g = 0; g < 8; ++g)
            sim[g] = fmaf(hp, keys[(g * 32 + lane) * PDIM + p], sim[g]);
    }

    float tv[TOPK]; int ti[TOPK];
    for (int k = 0; k < TOPK; ++k) {
        float bv = -__builtin_inff(); int bg = 0;
        #pragma unroll
        for (int g = 0; g < 8; ++g)
            if (sim[g] > bv) { bv = sim[g]; bg = g; }
        int bi = bg * 32 + lane;
        for (int off = 16; off; off >>= 1) {
            float ov = __shfl_xor(bv, off, 32);
            int   oi = __shfl_xor(bi, off, 32);
            if (ov > bv || (ov == bv && oi < bi)) { bv = ov; bi = oi; }
        }
        tv[k] = bv; ti[k] = bi;
        if (lane == (bi & 31)) sim[bi >> 5] = -__builtin_inff();
    }

    float e0 = 1.f;
    float e1 = __expf(tv[1] - tv[0]);
    float e2 = __expf(tv[2] - tv[0]);
    float e3 = __expf(tv[3] - tv[0]);
    float inv = 1.f / (e0 + e1 + e2 + e3);
    if (lane == 0) {
        tidx[t * 4 + 0] = ti[0]; tw[t * 4 + 0] = e0 * inv;
        tidx[t * 4 + 1] = ti[1]; tw[t * 4 + 1] = e1 * inv;
        tidx[t * 4 + 2] = ti[2]; tw[t * 4 + 2] = e2 * inv;
        tidx[t * 4 + 3] = ti[3]; tw[t * 4 + 3] = e3 * inv;
    }
}

// ---------------- 3) gathered pattern apply (one wave32 per token) --------
__global__ __launch_bounds__(256) void apply_kernel(
    float* __restrict__ x,
    const float* __restrict__ vd /*[256,1024,32]*/,
    const float* __restrict__ vu /*[256,32,1024]*/,
    const int* __restrict__ tidx, const float* __restrict__ tw,
    const float* __restrict__ scale_l)
{
    int lane = threadIdx.x & 31;
    int wave = threadIdx.x >> 5;
    int t = blockIdx.x * 8 + wave;
    float* xr = x + (size_t)t * DD;
    float sc = *scale_l;

    float acc[32];
    #pragma unroll
    for (int i = 0; i < 32; ++i) acc[i] = 0.f;

    for (int kk = 0; kk < TOPK; ++kk) {
        int idx = tidx[t * 4 + kk];
        float wk = tw[t * 4 + kk];
        const float* vdm = vd + (size_t)idx * DD * PDIM;
        float proj = 0.f;
        #pragma unroll 4
        for (int d = 0; d < DD; ++d)
            proj = fmaf(xr[d], vdm[d * PDIM + lane], proj);
        proj = proj / (1.f + __expf(-proj)) * wk;   // silu * softmax weight
        const float* vum = vu + (size_t)idx * PDIM * DD;
        for (int p = 0; p < PDIM; ++p) {
            float pp = __shfl(proj, p, 32);
            const float* row = vum + p * DD;
            #pragma unroll
            for (int i = 0; i < 32; ++i)
                acc[i] = fmaf(pp, row[i * 32 + lane], acc[i]);
        }
    }
    #pragma unroll
    for (int i = 0; i < 32; ++i) {
        int d = i * 32 + lane;
        xr[d] = xr[d] + acc[i] * sc;
    }
}

// ---------------- 4) RMSNorm + cast to bf16 ----------------
__global__ __launch_bounds__(256) void rmsnorm_kernel(
    const float* __restrict__ x, const float* __restrict__ norm_w,
    unsigned short* __restrict__ xb)
{
    __shared__ float red[8];
    int t = blockIdx.x;
    const float* xr = x + (size_t)t * DD;
    float s = 0.f;
    for (int d = threadIdx.x; d < DD; d += 256) { float v = xr[d]; s = fmaf(v, v, s); }
    for (int off = 16; off; off >>= 1) s += __shfl_xor(s, off, 32);
    int lane = threadIdx.x & 31, wave = threadIdx.x >> 5;
    if (lane == 0) red[wave] = s;
    __syncthreads();
    float tot = 0.f;
    #pragma unroll
    for (int w = 0; w < 8; ++w) tot += red[w];
    float rs = rsqrtf(tot * (1.f / DD) + EPS_RMS);
    for (int d = threadIdx.x; d < DD; d += 256)
        xb[(size_t)t * DD + d] = f2bf(xr[d] * rs * norm_w[d]);
}

// ---------------- 5) lm_head weights -> bf16 ----------------
__global__ __launch_bounds__(256) void cvt_kernel(
    const float* __restrict__ w, unsigned short* __restrict__ wb, int n)
{
    int i = blockIdx.x * 256 + threadIdx.x;
    if (i < n) wb[i] = f2bf(w[i]);
}

// ---------------- 6) LM head GEMM: TDM -> LDS -> WMMA bf16 ----------------
// C[4096,32000] = Xb[4096,1024] * Wb[32000,1024]^T  (both K-major)
// Block tile 128M x 128N, 8 waves = 4M x 2N, wave tile 32M x 64N.
// Per 32-K step: TDM loads A tile (128x32) and B tile (128x32) into LDS,
// double buffered; tiles have 16B row padding -> 80B row stride.

#define TILE_BYTES 10240            // 128 rows * 80 B
#define ROW_US     40               // padded row stride in ushorts

// Build a 2D tensor D# and issue TENSOR_LOAD_TO_LDS.
//  tile: 32 elems (K, contiguous) x 128 rows; element = 2 bytes (bf16)
//  LDS padding: every 16 DWORDs (64B) insert 4 DWORDs (16B)  -> 80B stride
static __device__ __forceinline__ void tdm_load_tile(
    const void* lds_ptr, const unsigned short* gptr, unsigned nrows)
{
    // low 32 bits of a generic LDS address = LDS byte offset (aperture rule)
    unsigned lds_off = (unsigned)(size_t)lds_ptr;
    unsigned long long ga = (unsigned long long)(size_t)gptr;
    u32x4 g0;
    g0[0] = 1u;                                            // count=1 (valid user D#)
    g0[1] = lds_off;                                       // lds_addr
    g0[2] = (unsigned)(ga & 0xFFFFFFFFu);                  // global_addr[31:0]
    g0[3] = (unsigned)((ga >> 32) & 0x01FFFFFFu) | (2u << 30);  // addr[56:32] | type=2
    i32x8 g1;
    g1[0] = (1 << 16) | (1 << 20) | (3 << 22) | (3 << 25); // data_size=2B, pad_en, ivl=16DW, amt=4DW
    g1[1] = (int)((DD & 0xFFFF) << 16);                    // tensor_dim0 lo16 (=1024)
    g1[2] = (int)((DD >> 16) & 0xFFFF) | (int)((nrows & 0xFFFFu) << 16); // dim0 hi | dim1 lo
    g1[3] = (int)((nrows >> 16) & 0xFFFFu) | (32 << 16);   // dim1 hi | tile_dim0=32
    g1[4] = 128;                                           // tile_dim1=128, tile_dim2=0
    g1[5] = DD;                                            // tensor_dim0_stride lo32 (=1024 elems)
    g1[6] = 0;                                             // stride hi | dim1_stride lo
    g1[7] = 0;
    i32x4 z4 = {0, 0, 0, 0};
#if defined(__clang_major__) && (__clang_major__ >= 23)
    i32x8 z8 = {0, 0, 0, 0, 0, 0, 0, 0};
    __builtin_amdgcn_tensor_load_to_lds(g0, g1, z4, z4, z8, 0);
#else
    __builtin_amdgcn_tensor_load_to_lds(g0, g1, z4, z4, 0);
#endif
}

// Plain (non-volatile) LDS reads -> ds_load_b128.  The visible init stores in
// the kernel prologue plus the seq-cst workgroup fence inside __syncthreads()
// keep these loads from being folded or forwarded across k-steps.
static __device__ __forceinline__ v16bf lds_a_frag(const unsigned short* base, int row, int hi8) {
    const unsigned short* p = base + row * ROW_US;
    ushort8 a = *(const ushort8*)(p + hi8);        // K hi8..hi8+7
    ushort8 b = *(const ushort8*)(p + 16 + hi8);   // K 16+hi8..23+hi8
    v16u r;
    #pragma unroll
    for (int i = 0; i < 8; ++i) { r[i] = a[i]; r[i + 8] = b[i]; }
    return __builtin_bit_cast(v16bf, r);
}
static __device__ __forceinline__ v16bf lds_b_frag(const unsigned short* base, int col, int hi) {
    const unsigned short* p = base + col * ROW_US + hi * 8;    // K 8hi..8hi+7, 16+8hi..23+8hi
    ushort8 a = *(const ushort8*)p;
    ushort8 b = *(const ushort8*)(p + 16);
    v16u r;
    #pragma unroll
    for (int i = 0; i < 8; ++i) { r[i] = a[i]; r[i + 8] = b[i]; }
    return __builtin_bit_cast(v16bf, r);
}

__global__ __launch_bounds__(256) void lm_head_kernel(
    const unsigned short* __restrict__ Xb,  // [4096,1024] bf16
    const unsigned short* __restrict__ Wb,  // [32000,1024] bf16
    float* __restrict__ out)                // [4096,32000] f32
{
    __shared__ __align__(16) unsigned char smem[4 * TILE_BYTES]; // A0 A1 B0 B1

    int lane = threadIdx.x & 31;
    int wave = threadIdx.x >> 5;
    int mw = wave >> 1;            // 0..3
    int nw = wave & 1;             // 0..1
    int col = lane & 15;
    int hi  = lane >> 4;           // 0/1
    int hi8 = hi << 3;

    // Visible may-alias stores over all of smem: prevents the compiler from
    // treating LDS (written for real by the TDM) as undef and deleting the
    // fragment loads.  Value is non-constant so loads can't be const-folded.
    {
        int* si = (int*)smem;
        for (int i = threadIdx.x; i < (4 * TILE_BYTES) / 4; i += 256)
            si[i] = (int)threadIdx.x;
    }
    __syncthreads();

    int mblk = blockIdx.y * 128;
    int nblk = blockIdx.x * 128;
    const unsigned short* Ag = Xb + (size_t)mblk * DD;
    const unsigned short* Bg = Wb + (size_t)nblk * DD;

    v8f acc[2][4];
    #pragma unroll
    for (int s = 0; s < 2; ++s)
        #pragma unroll
        for (int j = 0; j < 4; ++j) acc[s][j] = (v8f){};

    const int KSTEPS = DD / 32;    // 32

    // prologue: fetch tiles for k-step 0
    if (wave == 0) {
        tdm_load_tile(smem + 0 * TILE_BYTES, Ag + 0, NTOK);
        tdm_load_tile(smem + 2 * TILE_BYTES, Bg + 0, VV);
    }

    for (int ks = 0; ks < KSTEPS; ++ks) {
        if (wave == 0) {
            if (ks + 1 < KSTEPS) {
                int k1 = (ks + 1) * 32;
                tdm_load_tile(smem + ((ks + 1) & 1) * TILE_BYTES, Ag + k1, NTOK);
                tdm_load_tile(smem + (2 + ((ks + 1) & 1)) * TILE_BYTES, Bg + k1, VV);
                // two newest in flight; the two for this step must be done
                __builtin_amdgcn_s_wait_tensorcnt(2);
            } else {
                __builtin_amdgcn_s_wait_tensorcnt(0);
            }
        }
        __syncthreads();   // tiles for this step visible to all waves

        const unsigned short* At = (const unsigned short*)(smem + (ks & 1) * TILE_BYTES);
        const unsigned short* Bt = (const unsigned short*)(smem + (2 + (ks & 1)) * TILE_BYTES);

        v16bf a0 = lds_a_frag(At, mw * 32 + col, hi8);
        v16bf a1 = lds_a_frag(At, mw * 32 + 16 + col, hi8);
        #pragma unroll
        for (int j = 0; j < 4; ++j) {
            v16bf bf = lds_b_frag(Bt, nw * 64 + j * 16 + col, hi);
            acc[0][j] = __builtin_amdgcn_wmma_f32_16x16x32_bf16(false, a0, false, bf, (short)0, acc[0][j], false, false);
            acc[1][j] = __builtin_amdgcn_wmma_f32_16x16x32_bf16(false, a1, false, bf, (short)0, acc[1][j], false, false);
        }
        __syncthreads();   // done reading before buffers are overwritten
    }

    // C layout: VGPR v -> row (tilebase + v + 8*hi), col (base + lane&15)
    #pragma unroll
    for (int s = 0; s < 2; ++s) {
        int mbase = mblk + mw * 32 + s * 16 + 8 * hi;
        #pragma unroll
        for (int v = 0; v < 8; ++v) {
            float* o = out + (size_t)(mbase + v) * VV + nblk + nw * 64 + col;
            o[0]  = acc[s][0][v];
            o[16] = acc[s][1][v];
            o[32] = acc[s][2][v];
            o[48] = acc[s][3][v];
        }
    }
}

// ---------------- launcher ----------------
extern "C" void kernel_launch(void* const* d_in, const int* in_sizes, int n_in,
                              void* d_out, int out_size, void* d_ws, size_t ws_size,
                              hipStream_t stream) {
    (void)in_sizes; (void)n_in; (void)out_size; (void)ws_size;
    const int*   tokens    = (const int*)d_in[0];
    const float* embed     = (const float*)d_in[1];
    const float* lm_head_w = (const float*)d_in[2];
    const float* norm_w    = (const float*)d_in[3];
    const float* keys      = (const float*)d_in[4];
    const float* v_down    = (const float*)d_in[5];
    const float* v_up      = (const float*)d_in[6];
    const float* hasher_w  = (const float*)d_in[7];
    const float* scale     = (const float*)d_in[8];
    float* out = (float*)d_out;

    char* ws = (char*)d_ws;
    float*          x    = (float*)ws;                                  // 16 MB
    unsigned short* xb   = (unsigned short*)(ws + 16777216);            // 8 MB
    unsigned short* wb   = (unsigned short*)(ws + 25165824);            // 65.5 MB
    int*            tidx = (int*)(ws + 90701824);                       // 64 KB
    float*          tw   = (float*)(ws + 90767360);                     // 64 KB

    embed_kernel<<<NTOK, 256, 0, stream>>>(tokens, embed, x);

    for (int l = 0; l < LL; ++l) {
        hasher_topk_kernel<<<NTOK / 8, 256, 0, stream>>>(
            x, hasher_w + (size_t)l * PDIM * DD, keys + (size_t)l * NPAT * PDIM,
            tidx, tw);
        apply_kernel<<<NTOK / 8, 256, 0, stream>>>(
            x, v_down + (size_t)l * NPAT * DD * PDIM,
            v_up + (size_t)l * NPAT * PDIM * DD, tidx, tw, scale + l);
    }

    rmsnorm_kernel<<<NTOK, 256, 0, stream>>>(x, norm_w, xb);

    int nw = VV * DD;  // 32,768,000
    cvt_kernel<<<(nw + 255) / 256, 256, 0, stream>>>(lm_head_w, wb, nw);

    // grid: N tiles fastest -> whole W streams per 128-row M band, stays L2-hot
    lm_head_kernel<<<dim3(VV / 128, NTOK / 128), 256, 0, stream>>>(xb, wb, out);
}
